// DSAttention_17265768530055
// MI455X (gfx1250) — compile-verified, matrix operand order checked by reading, not run on hardware
//
#include <hip/hip_runtime.h>

typedef _Float16 half_t;
typedef __attribute__((ext_vector_type(16))) _Float16 v16h;
typedef __attribute__((ext_vector_type(8)))  _Float16 v8h;
typedef __attribute__((ext_vector_type(8)))  float    v8f;
typedef __attribute__((ext_vector_type(4)))  float    v4f;
typedef int v4i __attribute__((vector_size(16)));   // matches async-LDS builtin param

#define HWSZ 65536
#define NWIN 8192        // 8 * 32 * 32

#define AS1 __attribute__((address_space(1)))
#define AS3 __attribute__((address_space(3)))

#if __has_builtin(__builtin_amdgcn_global_load_async_to_lds_b128)
#define HAVE_ASYNC_LDS 1
#else
#define HAVE_ASYNC_LDS 0
#endif

// ---------------- WMMA helpers ----------------

__device__ __forceinline__ v8f wmma16(v16h a, v16h b, v8f c) {
  return __builtin_amdgcn_wmma_f32_16x16x32_f16(
      /*neg_a=*/false, a, /*neg_b=*/false, b,
      /*c_mod=*/(short)0, c, /*reuse_a=*/false, /*reuse_b=*/false);
}

// Fragment load for A (row-major) or B (source stored [col][k] row-major).
// koff = k0 + ((lane&16)?8:0). halves 0..7 = K[koff..koff+7],
// halves 8..15 = K[koff+16..koff+23] (CDNA5 ISA 16-bit A 16x32 layout).
__device__ __forceinline__ v16h frag_ld(const half_t* rowbase, int koff) {
  v8h lo = *(const v8h*)(rowbase + koff);
  v8h hi = *(const v8h*)(rowbase + koff + 16);
  v16h r;
#pragma unroll
  for (int j = 0; j < 8; ++j) { r[j] = lo[j]; r[j + 8] = hi[j]; }
  return r;
}

__device__ __forceinline__ void wait_async0() {
#if __has_builtin(__builtin_amdgcn_s_wait_asynccnt)
  __builtin_amdgcn_s_wait_asynccnt(0);
#elif HAVE_ASYNC_LDS
  asm volatile("s_wait_asynccnt 0" ::: "memory");
#endif
}

// ---------------- Kernel 0: prep (weights -> f16, rel-pos MLP bias) ----------------

__global__ void k_prep(const float* __restrict__ Wq, const float* __restrict__ Wkv,
                       const float* __restrict__ Wproj,
                       const float* __restrict__ mw1, const float* __restrict__ mb1,
                       const float* __restrict__ mw2, const float* __restrict__ mb2,
                       half_t* __restrict__ Wqkv3, half_t* __restrict__ Wprojh,
                       float* __restrict__ biasb) {
  int t = blockIdx.x * 256 + threadIdx.x;
  if (t < 12288) {                    // bias MLP: [head][i][j]
    int head = t >> 12;
    int ij = t & 4095;
    int i = ij >> 6, j = ij & 63;
    float dy = (float)((i >> 3) - (j >> 3));
    float dx = (float)((i & 7) - (j & 7));
    float sy = (dy > 0.f) ? 1.f : ((dy < 0.f) ? -1.f : 0.f);
    float sx = (dx > 0.f) ? 1.f : ((dx < 0.f) ? -1.f : 0.f);
    float ry = sy * logf(1.f + fabsf(dy));
    float rx = sx * logf(1.f + fabsf(dx));
    float acc = mb2[head];
    for (int h = 0; h < 256; ++h) {
      float hv = mw1[h * 2] * ry + mw1[h * 2 + 1] * rx + mb1[h];
      hv = fmaxf(hv, 0.f);
      acc += mw2[head * 256 + h] * hv;
    }
    biasb[head * 4096 + ij] = acc;
  } else if (t < 12288 + 27648) {     // Wqkv3 f16 [288][96]
    int u = t - 12288;
    int o = u / 96, c = u - o * 96;
    float w = (o < 96) ? Wq[o * 96 + c] : Wkv[(o - 96) * 96 + c];
    Wqkv3[u] = (half_t)w;
  } else {                            // Wprojh f16 [96][96]
    int u = t - 39936;
    Wprojh[u] = (half_t)Wproj[u];
  }
}

// ---------------- Kernel 1: QKV 1x1-conv GEMM + window scatter ----------------
// Block: 256 threads (8 waves). Tile: M=128 pixels (contiguous in w), N=288, K=96.
// n-tiles split into uniform Q / K / V segments -> branch-free vector epilogues.

__global__ void k_qkv(const float* __restrict__ X,
                      const half_t* __restrict__ Wqkv3,
                      const float* __restrict__ bq, const float* __restrict__ bkv,
                      const float* __restrict__ logit_scale,
                      float* __restrict__ Qbuf, half_t* __restrict__ Kw,
                      half_t* __restrict__ Vt) {
  __shared__ __align__(16) half_t sA[128 * 104];

  int bid = blockIdx.x;
  int w0 = (bid & 1) * 128;
  int h  = (bid >> 1) & 255;
  int b  = bid >> 9;

  for (int idx = threadIdx.x; idx < 96 * 128; idx += 256) {
    int c = idx >> 7;
    int i = idx & 127;
    float xv = X[(((b * 96 + c) << 16)) + (h << 8) + w0 + i];
    sA[i * 104 + c] = (half_t)xv;
  }
  __builtin_prefetch(Wqkv3, 0, 3);
  __syncthreads();

  int wave = threadIdx.x >> 5;
  int lane = threadIdx.x & 31;
  int l15  = lane & 15;
  int hi8  = (lane & 16) ? 8 : 0;
  int m0   = wave * 16;

  float ls = expf(fminf(logit_scale[0], 4.6051702f));
  float kscale = ls * 0.17677669529663687f;   // ls / sqrt(HD)

  const half_t* arow = &sA[(m0 + l15) * 104];
  v16h afrag[3];
  afrag[0] = frag_ld(arow, 0 + hi8);
  afrag[1] = frag_ld(arow, 32 + hi8);
  afrag[2] = frag_ld(arow, 64 + hi8);

  // window coords for this lane's 8 contiguous rows (wb is a multiple of 8)
  int wb   = w0 + m0 + hi8;
  int winb = ((b * 32 + (h >> 3)) << 5) + (wb >> 3);
  int nrow = (h & 7) << 3;

  // ---- Q segment (cout 0..95): contiguous f32 stores ----
#pragma unroll 1
  for (int nt = 0; nt < 6; ++nt) {
    int n0 = nt << 4;
    const half_t* brow = Wqkv3 + (n0 + l15) * 96;
    v8f acc = {};
    acc = wmma16(afrag[0], frag_ld(brow, 0 + hi8), acc);
    acc = wmma16(afrag[1], frag_ld(brow, 32 + hi8), acc);
    acc = wmma16(afrag[2], frag_ld(brow, 64 + hi8), acc);
    int cout = n0 + l15;
    float bias = bq[cout];
    float* qd = Qbuf + ((b * 96 + cout) << 16) + (h << 8) + wb;
    v4f lo = {acc[0] + bias, acc[1] + bias, acc[2] + bias, acc[3] + bias};
    v4f hi = {acc[4] + bias, acc[5] + bias, acc[6] + bias, acc[7] + bias};
    *(v4f*)qd = lo;
    *(v4f*)(qd + 4) = hi;
  }

  // ---- K segment (cout 96..191): scaled, token-strided b16 stores ----
#pragma unroll 1
  for (int nt = 6; nt < 12; ++nt) {
    int n0 = nt << 4;
    const half_t* brow = Wqkv3 + (n0 + l15) * 96;
    v8f acc = {};
    acc = wmma16(afrag[0], frag_ld(brow, 0 + hi8), acc);
    acc = wmma16(afrag[1], frag_ld(brow, 32 + hi8), acc);
    acc = wmma16(afrag[2], frag_ld(brow, 64 + hi8), acc);
    int ck = (n0 - 96) + l15;
    int head = ck >> 5, hd = ck & 31;
    float bias = bkv[ck];
    half_t* kd = Kw + ((((winb * 3 + head) << 6) + nrow) << 5) + hd;
#pragma unroll
    for (int r = 0; r < 8; ++r)
      kd[r << 5] = (half_t)((acc[r] + bias) * kscale);
  }

  // ---- V segment (cout 192..287): contiguous v8h store (token index!) ----
#pragma unroll 1
  for (int nt = 12; nt < 18; ++nt) {
    int n0 = nt << 4;
    const half_t* brow = Wqkv3 + (n0 + l15) * 96;
    v8f acc = {};
    acc = wmma16(afrag[0], frag_ld(brow, 0 + hi8), acc);
    acc = wmma16(afrag[1], frag_ld(brow, 32 + hi8), acc);
    acc = wmma16(afrag[2], frag_ld(brow, 64 + hi8), acc);
    int cv = (n0 - 192) + l15;
    int head = cv >> 5, hd = cv & 31;
    float bias = bkv[cv + 96];
    half_t* vd = Vt + ((((winb * 3 + head) << 5) + hd) << 6) + nrow;
    v8h pv;
#pragma unroll
    for (int r = 0; r < 8; ++r) pv[r] = (half_t)(acc[r] + bias);
    *(v8h*)vd = pv;
  }
}

// ---------------- Kernel 2: depthwise 5x5 reflect conv -> window-layout f16 q ----------------

__global__ void k_dwconv(const float* __restrict__ Qbuf,
                         const float* __restrict__ dww, const float* __restrict__ dwb,
                         half_t* __restrict__ Qw) {
  __shared__ float st[20 * 20];
  int bid  = blockIdx.x;
  int tile = bid & 255;
  int bc   = bid >> 8;               // b*96 + c
  int b    = bc / 96;
  int c    = bc - b * 96;
  int ty0  = (tile >> 4) << 4;
  int tx0  = (tile & 15) << 4;

  for (int i = threadIdx.x; i < 400; i += 256) {
    int ly = i / 20, lx = i - ly * 20;
    int gy = ty0 + ly - 2, gx = tx0 + lx - 2;
    gy = (gy < 0) ? -gy : ((gy > 255) ? 510 - gy : gy);
    gx = (gx < 0) ? -gx : ((gx > 255) ? 510 - gx : gx);
    st[i] = Qbuf[(bc << 16) + (gy << 8) + gx];
  }
  __syncthreads();

  int ly = threadIdx.x >> 4, lx = threadIdx.x & 15;
  float s = dwb[c];
#pragma unroll
  for (int ky = 0; ky < 5; ++ky)
#pragma unroll
    for (int kx = 0; kx < 5; ++kx)
      s += dww[c * 25 + ky * 5 + kx] * st[(ly + ky) * 20 + (lx + kx)];

  int hh = ty0 + ly, w = tx0 + lx;
  int win = ((b * 32 + (hh >> 3)) << 5) + (w >> 3);
  int n   = ((hh & 7) << 3) + (w & 7);
  int head = c >> 5, hd = c & 31;
  Qw[((((win * 3 + head) << 6) + n) << 5) + hd] = (half_t)s;
}

// ---------------- Kernel 3: windowed attention (one block per window*head) ----------------
// Block: 128 threads (4 waves). q/k/v staged to LDS via async copy when available.

__global__ void k_attn(const half_t* __restrict__ Qw, const half_t* __restrict__ Kw,
                       const half_t* __restrict__ Vt, const float* __restrict__ biasb,
                       half_t* __restrict__ Xa) {
  __shared__ __align__(16) half_t sQ[2048];
  __shared__ __align__(16) half_t sK[2048];
  __shared__ __align__(16) half_t sV[2048];
  __shared__ __align__(16) float  SbufT[64 * 72];   // transposed: [key][query]
  __shared__ __align__(16) half_t Pbuf[64 * 72];    // [query][key]

  int task = blockIdx.x;
  int win  = task / 3;
  int head = task - win * 3;
  size_t base = (size_t)(win * 3 + head) << 11;
  const half_t* qb = Qw + base;
  const half_t* kb = Kw + base;
  const half_t* vb = Vt + base;
  const float*  brow = biasb + head * 4096;

  int t = threadIdx.x;
#if HAVE_ASYNC_LDS
  {
    // each buffer is 4 KB = 256 x b128; 128 threads x 2 transfers each
    AS1 v4i* gq = (AS1 v4i*)(qb);
    AS1 v4i* gk = (AS1 v4i*)(kb);
    AS1 v4i* gv = (AS1 v4i*)(vb);
    AS3 v4i* lq = (AS3 v4i*)sQ;
    AS3 v4i* lk = (AS3 v4i*)sK;
    AS3 v4i* lv = (AS3 v4i*)sV;
    __builtin_amdgcn_global_load_async_to_lds_b128(gq + t,       lq + t,       0, 0);
    __builtin_amdgcn_global_load_async_to_lds_b128(gq + t + 128, lq + t + 128, 0, 0);
    __builtin_amdgcn_global_load_async_to_lds_b128(gk + t,       lk + t,       0, 0);
    __builtin_amdgcn_global_load_async_to_lds_b128(gk + t + 128, lk + t + 128, 0, 0);
    __builtin_amdgcn_global_load_async_to_lds_b128(gv + t,       lv + t,       0, 0);
    __builtin_amdgcn_global_load_async_to_lds_b128(gv + t + 128, lv + t + 128, 0, 0);
    wait_async0();
  }
#else
  {
    v8h* dq = (v8h*)sQ; const v8h* gq = (const v8h*)qb;
    v8h* dk = (v8h*)sK; const v8h* gk = (const v8h*)kb;
    v8h* dv = (v8h*)sV; const v8h* gv = (const v8h*)vb;
    dq[t] = gq[t]; dq[t + 128] = gq[t + 128];
    dk[t] = gk[t]; dk[t + 128] = gk[t + 128];
    dv[t] = gv[t]; dv[t + 128] = gv[t + 128];
  }
#endif
  __syncthreads();

  int wave = t >> 5;
  int lane = t & 31;
  int l15  = lane & 15;
  int hi8  = (lane & 16) ? 8 : 0;
  int m0   = wave * 16;

  // S = q @ k^T  (K dim = HD = 32 -> one WMMA per 16x16 tile)
  v16h qf = frag_ld(sQ + (m0 + l15) * 32, hi8);
#pragma unroll
  for (int nt = 0; nt < 4; ++nt) {
    v8f a = {};
    a = wmma16(qf, frag_ld(sK + ((nt << 4) + l15) * 32, hi8), a);
    // transposed spill: lane's 8 query rows are contiguous -> two b128 stores
    float* sd = &SbufT[((nt << 4) + l15) * 72 + m0 + hi8];
    v4f lo = {a[0], a[1], a[2], a[3]};
    v4f hi = {a[4], a[5], a[6], a[7]};
    *(v4f*)sd = lo;
    *(v4f*)(sd + 4) = hi;
  }
  __syncthreads();

  // softmax: one thread per query row (threads 0..63); native v_exp_f32
  if (t < 64) {
    int m = t;
    float rowv[64];
    float mx = -1e30f;
#pragma unroll
    for (int j = 0; j < 64; ++j) {
      float v = SbufT[j * 72 + m] + brow[(m << 6) + j];
      rowv[j] = v;
      mx = fmaxf(mx, v);
    }
    float sum = 0.f;
#pragma unroll
    for (int j = 0; j < 64; ++j) {
      float e = __expf(rowv[j] - mx);   // range-reduced by row max -> fast exp is safe
      rowv[j] = e;
      sum += e;
    }
    float inv = 1.f / sum;
#pragma unroll
    for (int j8 = 0; j8 < 8; ++j8) {
      v8h p;
#pragma unroll
      for (int e = 0; e < 8; ++e) p[e] = (half_t)(rowv[j8 * 8 + e] * inv);
      *(v8h*)&Pbuf[m * 72 + (j8 << 3)] = p;
    }
  }
  __syncthreads();

  // O = P @ V   (K = 64 keys, N = 32 hd). sV stored [hd][key].
  v16h pf0 = frag_ld(&Pbuf[(m0 + l15) * 72], 0 + hi8);
  v16h pf1 = frag_ld(&Pbuf[(m0 + l15) * 72], 32 + hi8);
  int gp0 = win << 6;
#pragma unroll
  for (int nt = 0; nt < 2; ++nt) {
    const half_t* vrow = sV + ((nt << 4) + l15) * 64;
    v8f a = {};
    a = wmma16(pf0, frag_ld(vrow, 0 + hi8), a);
    a = wmma16(pf1, frag_ld(vrow, 32 + hi8), a);
    int cch = head * 32 + (nt << 4) + l15;
#pragma unroll
    for (int r = 0; r < 8; ++r) {
      int m = m0 + r + hi8;
      Xa[(size_t)(gp0 + m) * 96 + cch] = (half_t)a[r];
    }
  }
}

// ---------------- Kernel 4: output projection GEMM + window reverse ----------------
// Block: 256 threads (8 waves). Tile: M=128 window-ordered pixels, N=96, K=96.

__global__ void k_proj(const half_t* __restrict__ Xa, const half_t* __restrict__ Wprojh,
                       const float* __restrict__ bproj, float* __restrict__ out) {
  int wave = threadIdx.x >> 5;
  int lane = threadIdx.x & 31;
  int l15  = lane & 15;
  int hi8  = (lane & 16) ? 8 : 0;
  int gp0  = blockIdx.x * 128 + wave * 16;

  __builtin_prefetch(Wprojh, 0, 3);

  const half_t* arow = Xa + (size_t)(gp0 + l15) * 96;
  v16h afrag[3];
  afrag[0] = frag_ld(arow, 0 + hi8);
  afrag[1] = frag_ld(arow, 32 + hi8);
  afrag[2] = frag_ld(arow, 64 + hi8);

  // lane's 8 output rows are 8 contiguous w within one window row
  int gpb = gp0 + hi8;                   // multiple of 8
  int win = gpb >> 6, nb = gpb & 63;
  int bI  = win >> 10;
  int wy  = (win >> 5) & 31;
  int wx  = win & 31;
  int hh  = (wy << 3) + (nb >> 3);
  int wcol = (wx << 3);

#pragma unroll 1
  for (int nt = 0; nt < 6; ++nt) {
    int n0 = nt << 4;
    const half_t* brow = Wprojh + (n0 + l15) * 96;
    v8f acc = {};
    acc = wmma16(afrag[0], frag_ld(brow, 0 + hi8), acc);
    acc = wmma16(afrag[1], frag_ld(brow, 32 + hi8), acc);
    acc = wmma16(afrag[2], frag_ld(brow, 64 + hi8), acc);

    int cout = n0 + l15;
    float bias = bproj[cout];
    float* od = out + ((bI * 96 + cout) << 16) + (hh << 8) + wcol;
    v4f lo = {acc[0] + bias, acc[1] + bias, acc[2] + bias, acc[3] + bias};
    v4f hi = {acc[4] + bias, acc[5] + bias, acc[6] + bias, acc[7] + bias};
    *(v4f*)od = lo;
    *(v4f*)(od + 4) = hi;
  }
}

// ---------------- launch ----------------

extern "C" void kernel_launch(void* const* d_in, const int* in_sizes, int n_in,
                              void* d_out, int out_size, void* d_ws, size_t ws_size,
                              hipStream_t stream) {
  const float* X      = (const float*)d_in[0];
  const float* Wq     = (const float*)d_in[1];
  const float* bq     = (const float*)d_in[2];
  const float* Wkv    = (const float*)d_in[3];
  const float* bkv    = (const float*)d_in[4];
  const float* dww    = (const float*)d_in[5];
  const float* dwb    = (const float*)d_in[6];
  const float* Wproj  = (const float*)d_in[7];
  const float* bproj  = (const float*)d_in[8];
  const float* lscale = (const float*)d_in[9];
  const float* mw1    = (const float*)d_in[10];
  const float* mb1    = (const float*)d_in[11];
  const float* mw2    = (const float*)d_in[12];
  const float* mb2    = (const float*)d_in[13];
  float* out = (float*)d_out;

  char* ws = (char*)d_ws;
  size_t o = 0;
  half_t* Wqkv3  = (half_t*)(ws + o); o += 288 * 96 * sizeof(half_t);
  half_t* Wprojh = (half_t*)(ws + o); o += 96 * 96 * sizeof(half_t);
  float*  biasb  = (float*)(ws + o);  o += 3 * 64 * 64 * sizeof(float);
  o = (o + 255) & ~(size_t)255;
  float*  Qbuf = (float*)(ws + o);  o += (size_t)768 * HWSZ * sizeof(float);        // 192 MiB
  half_t* Qw   = (half_t*)(ws + o); o += (size_t)NWIN * 3 * 2048 * sizeof(half_t);  // 96 MiB
  half_t* Kw   = (half_t*)(ws + o); o += (size_t)NWIN * 3 * 2048 * sizeof(half_t);  // 96 MiB
  half_t* Vt   = (half_t*)(ws + o); o += (size_t)NWIN * 3 * 2048 * sizeof(half_t);  // 96 MiB
  half_t* Xa   = (half_t*)(ws + o); o += (size_t)NWIN * 64 * 96 * sizeof(half_t);   // 96 MiB

  k_prep<<<192, 256, 0, stream>>>(Wq, Wkv, Wproj, mw1, mb1, mw2, mb2,
                                  Wqkv3, Wprojh, biasb);
  k_qkv<<<4096, 256, 0, stream>>>(X, Wqkv3, bq, bkv, lscale, Qbuf, Kw, Vt);
  k_dwconv<<<196608, 256, 0, stream>>>(Qbuf, dww, dwb, Qw);
  k_attn<<<24576, 128, 0, stream>>>(Qw, Kw, Vt, biasb, Xa);
  k_proj<<<4096, 256, 0, stream>>>(Xa, Wprojh, bproj, out);
}
